// Decoder_20048907338082
// MI455X (gfx1250) — matrix operand order repaired and checked by reading, not compiled, not run on hardware
//
#include <hip/hip_runtime.h>
#include <hip/hip_bf16.h>
#include <stdint.h>

// Problem constants (from reference)
#define P_DIM   256
#define L_DIM   17
#define STEPS   (P_DIM * L_DIM)   // 4352 sequential LSTM steps
#define NPROV   81
#define HENC    120
#define EMB     80
#define LSTM    200
#define GATES   (4 * LSTM)        // 800
#define VOCAB   13042
#define XDIM    400               // [x(200), h(200)] combined for the cell GEMV

// WMMA tiling
#define KPAD        224           // 200 padded up to 7 * 32
#define KTILES      7
#define COLS_PER_WG 256
#define NWG         51            // 51 * 256 = 13056 >= 13042
#define COLS_PAD    (NWG * COLS_PER_WG)
#define THREADS     256

typedef __attribute__((ext_vector_type(16))) __bf16 v16bf;
typedef __attribute__((ext_vector_type(8)))  float  v8f;
typedef __attribute__((ext_vector_type(4)))  unsigned int u32x4;
typedef __attribute__((ext_vector_type(8)))  int i32x8;
typedef __attribute__((ext_vector_type(4)))  int i32x4;

// Tensor Data Mover availability (device pass only; host pass takes fallback)
#if defined(__HIP_DEVICE_COMPILE__) && __has_builtin(__builtin_amdgcn_tensor_load_to_lds)
#define USE_TDM 1
#else
#define USE_TDM 0
#endif

// ---------------- workspace layout (bytes) ----------------
#define OFF_WCM   0ull
#define SZ_WCM    ((unsigned long long)COLS_PAD * KPAD * 2)   // bf16 col-major dense_W: 5,849,088
#define OFF_WXHT  (OFF_WCM + SZ_WCM)                          // f32 [800][400] transposed [Wx;Wh]
#define SZ_WXHT   ((unsigned long long)GATES * XDIM * 4)      // 1,280,000
#define OFF_ENC   (OFF_WXHT + SZ_WXHT)                        // f32 [4352][120] gathered encoder rows
#define SZ_ENC    ((unsigned long long)STEPS * HENC * 4)      // 2,088,960
#define OFF_LMAX  (OFF_ENC + SZ_ENC)                          // per-WG local max
#define OFF_LSUM  (OFF_LMAX + 256)                            // per-WG local expsum
#define OFF_LARG  (OFF_LSUM + 256)                            // per-WG local argmax
#define OFF_H     (OFF_LARG + 256)                            // published h_new (200 f32)
#define OFF_SYNC  (OFF_H + 1024)
// total ~9.22 MB

struct SyncBlk {
  unsigned int arrive;   // monotonically increasing arrival counter
  int   flagH;           // h_new published for unmasked-step k  (value k+1)
  int   flagA;           // gmax/gsum/action published for step k (value k+1)
  int   gaction;
  float gmax;
  float gsum;
};

// LDS: 112KB weight slice + f32 scratch
#define SMEM_BYTES 123456
// bf16 elements per WG weight slice (= 0xE000, fits tile_dim0's 16 bits)
#define SLICE_ELEMS (COLS_PER_WG * KPAD)

__device__ __forceinline__ float leaky(float v) { return v > 0.0f ? v : 0.2f * v; }

// ---------------- prep kernels ----------------

// dense_W f32 row-major [200][13042] -> bf16 col-major padded [13056][224]
__global__ void prep_wcm_kernel(const float* __restrict__ dW, __bf16* __restrict__ wcm) {
  long long i = (long long)blockIdx.x * blockDim.x + threadIdx.x;
  long long n = (long long)COLS_PAD * KPAD;
  if (i >= n) return;
  int col = (int)(i / KPAD);
  int kk  = (int)(i % KPAD);
  float v = (col < VOCAB && kk < LSTM) ? dW[(long long)kk * VOCAB + col] : 0.0f;
  wcm[i] = (__bf16)v;
}

// [Wx;Wh] (400x800, row-major pieces) -> transposed f32 [800][400]
__global__ void prep_wxht_kernel(const float* __restrict__ Wx, const float* __restrict__ Wh,
                                 float* __restrict__ wxht) {
  int i = blockIdx.x * blockDim.x + threadIdx.x;
  if (i >= GATES * XDIM) return;
  int g = i / XDIM;
  int j = i % XDIM;
  wxht[i] = (j < LSTM) ? Wx[(long long)j * GATES + g] : Wh[(long long)(j - LSTM) * GATES + g];
}

// gather enc_flat[s][f] = h_enc[p][locations[p][l]][f]
__global__ void prep_enc_kernel(const float* __restrict__ h_enc, const int* __restrict__ loc,
                                float* __restrict__ encf) {
  int i = blockIdx.x * blockDim.x + threadIdx.x;
  if (i >= STEPS * HENC) return;
  int s = i / HENC;
  int f = i % HENC;
  int p = s / L_DIM;
  int l = s % L_DIM;
  int lc = loc[p * L_DIM + l];
  encf[i] = h_enc[((long long)p * NPROV + lc) * HENC + f];
}

// zero entire output (masked steps stay zero; unmasked rows are overwritten)
__global__ void prep_zero_kernel(float* __restrict__ out, long long n) {
  long long stride = (long long)gridDim.x * blockDim.x;
  for (long long i = (long long)blockIdx.x * blockDim.x + threadIdx.x; i < n; i += stride)
    out[i] = 0.0f;
}

__global__ void prep_sync_kernel(SyncBlk* __restrict__ sync, float* __restrict__ lmax,
                                 float* __restrict__ lsum, int* __restrict__ larg) {
  int t = threadIdx.x;
  if (t == 0) {
    sync->arrive = 0u; sync->flagH = 0; sync->flagA = 0;
    sync->gaction = 0; sync->gmax = 0.0f; sync->gsum = 1.0f;
  }
  if (t < NWG) { lmax[t] = 0.0f; lsum[t] = 0.0f; larg[t] = 0; }
}

// ---------------- persistent decode kernel ----------------
__global__ __launch_bounds__(THREADS)
void decode_kernel(const __bf16* __restrict__ wcm, const float* __restrict__ wxht,
                   const float* __restrict__ encf, const float* __restrict__ emb,
                   const float* __restrict__ b_lstm, const float* __restrict__ dense_b,
                   const int* __restrict__ mask, const int* __restrict__ go_id,
                   float* __restrict__ ws_lmax, float* __restrict__ ws_lsum,
                   int* __restrict__ ws_larg, float* __restrict__ ws_h,
                   SyncBlk* __restrict__ sync,
                   float* __restrict__ out_actions, float* __restrict__ out_probs) {
  extern __shared__ char smem[];
  __bf16* ldsW    = (__bf16*)smem;                                   // [256][224] bf16 = 112KB
  float*  s_logit = (float*)(smem + (size_t)COLS_PER_WG * KPAD * 2); // 256
  float*  s_hpad  = s_logit + 256;   // 224 (h_new zero-padded)
  float*  s_red   = s_hpad + 224;    // 16
  int*    s_redi  = (int*)(s_red + 16); // 16
  float*  s_xh    = (float*)(s_redi + 16); // 400 (cell input [a,e,h])
  float*  s_z     = s_xh + 400;      // 800 (gate preacts)
  float*  s_c     = s_z + 800;       // 200 (carry c, WG0 only)
  float*  s_a     = s_c + 200;       // 80  (carry action-embedding, WG0 only)
  float*  s_hst   = s_a + 80;        // 200 (carry h, WG0 only)

  const int wg   = blockIdx.x;
  const int tid  = threadIdx.x;
  const int lane = tid & 31;
  const int wv   = tid >> 5;

  // ---- Stage this WG's bf16 weight slice into LDS once (resident for all steps) ----
#if USE_TDM
  // Tensor Data Mover: 1-row 2D tile of 57344 x 2B elements, global -> LDS.
  // D# per ISA ch.8: group0 {count=1, lds_addr, global_addr, type=2},
  // group1 {data_size=1(2B), tensor_dim0=tile_dim0=0xE000, tensor_dim1=tile_dim1=1,
  //         tensor_dim0_stride=0xE000}, groups 2/3 zero (<=2D tensor).
  if (wv == 0) {
    unsigned long long ga =
        (unsigned long long)(const char*)(wcm + (size_t)wg * SLICE_ELEMS);
    unsigned ldsoff = (unsigned)__builtin_amdgcn_groupstaticsize(); // dynamic LDS base
    u32x4 g0;
    g0[0] = 1u;                                              // count=1, user mode
    g0[1] = ldsoff;                                          // lds_addr (bytes)
    g0[2] = (unsigned)(ga & 0xFFFFFFFFull);                  // global_addr[31:0]
    g0[3] = (unsigned)((ga >> 32) & 0x01FFFFFFull) | 0x80000000u; // addr[56:32] | type=2
    i32x8 g1;
    g1[0] = 0x00010000;        // workgroup_mask=0, data_size=1 (2 bytes)
    g1[1] = (int)0xE0000000;   // tensor_dim0[15:0]=0xE000 in bits[63:48]
    g1[2] = 0x00010000;        // tensor_dim0[31:16]=0, tensor_dim1[15:0]=1
    g1[3] = (int)0xE0000000;   // tensor_dim1[31:16]=0, tile_dim0=0xE000
    g1[4] = 0x00000001;        // tile_dim1=1, tile_dim2=0
    g1[5] = SLICE_ELEMS;       // tensor_dim0_stride[31:0]
    g1[6] = 0;                 // stride hi / tensor_dim1_stride lo
    g1[7] = 0;                 // tensor_dim1_stride hi
    i32x4 gz = {0, 0, 0, 0};
#if defined(__clang_major__) && (__clang_major__ >= 23)
    i32x8 gz8 = {0, 0, 0, 0, 0, 0, 0, 0};
    __builtin_amdgcn_tensor_load_to_lds(g0, g1, gz, gz, gz8, 0);
#else
    __builtin_amdgcn_tensor_load_to_lds(g0, g1, gz, gz, 0);
#endif
    __builtin_amdgcn_s_wait_tensorcnt(0);  // TENSORcnt==0: tile landed in LDS
  }
#else
  {
    const float4* src = (const float4*)(wcm + (size_t)wg * SLICE_ELEMS);
    float4* dst = (float4*)ldsW;
    const int n16 = SLICE_ELEMS * 2 / 16;
    for (int i = tid; i < n16; i += THREADS) dst[i] = src[i];
  }
#endif
  // WG0 carried state init: h0 = [emb[go], 0...], c0 = 0, a0 = 0
  if (wg == 0) {
    int go = go_id[0];
    for (int i = tid; i < LSTM; i += THREADS) {
      s_hst[i] = (i < EMB) ? emb[(size_t)go * EMB + i] : 0.0f;
      s_c[i]   = 0.0f;
    }
    for (int i = tid; i < EMB; i += THREADS) s_a[i] = 0.0f;
  }
  __syncthreads();

  int k = 0; // unmasked-step counter
  for (int s = 0; s < STEPS; ++s) {
    if (mask[s] == 0) continue;  // masked: outputs pre-zeroed, carry unchanged -> skip all work

    // ---- Stage 1 (WG0): LSTM cell in f32 -> publish h_new ----
    if (wg == 0) {
      if (tid == 0) __builtin_prefetch(encf + (size_t)s * HENC, 0, 3);
      for (int i = tid; i < XDIM; i += THREADS) {
        float v;
        if (i < EMB)            v = s_a[i];
        else if (i < EMB+HENC)  v = encf[(size_t)s * HENC + (i - EMB)];
        else                    v = s_hst[i - (EMB + HENC)];
        s_xh[i] = v;
      }
      __syncthreads();
      for (int g = tid; g < GATES; g += THREADS) {
        const float4* wr = (const float4*)(wxht + (size_t)g * XDIM);
        const float4* xr = (const float4*)s_xh;
        float acc = b_lstm[g];
        #pragma unroll 4
        for (int j = 0; j < XDIM / 4; ++j) {
          float4 w4 = wr[j]; float4 x4 = xr[j];
          acc += w4.x * x4.x + w4.y * x4.y + w4.z * x4.z + w4.w * x4.w;
        }
        s_z[g] = acc;
      }
      __syncthreads();
      for (int t = tid; t < LSTM; t += THREADS) {
        float zi = s_z[t], zf = s_z[LSTM + t], zg = s_z[2 * LSTM + t], zo = s_z[3 * LSTM + t];
        float ig = 1.0f / (1.0f + expf(-zi));
        float fg = 1.0f / (1.0f + expf(-zf));
        float og = 1.0f / (1.0f + expf(-zo));
        float cn = fg * s_c[t] + ig * leaky(zg);
        float hn = og * leaky(cn);
        s_c[t] = cn; s_hst[t] = hn;
        ws_h[t] = hn;
      }
      __threadfence();
      __syncthreads();
      if (tid == 0)
        __hip_atomic_store(&sync->flagH, k + 1, __ATOMIC_RELEASE, __HIP_MEMORY_SCOPE_AGENT);
    }

    // ---- Stage 2 (all WGs): wait for h_new, compute logits slice via WMMA ----
    if (tid == 0) {
      while (__hip_atomic_load(&sync->flagH, __ATOMIC_ACQUIRE, __HIP_MEMORY_SCOPE_AGENT) < k + 1)
        __builtin_amdgcn_s_sleep(1);
    }
    __syncthreads();
    __builtin_amdgcn_fence(__ATOMIC_ACQUIRE, "agent");

    for (int i = tid; i < KPAD; i += THREADS)
      s_hpad[i] = (i < LSTM) ? ws_h[i] : 0.0f;
    __syncthreads();

    // A fragments: 16x32 bf16 tile with every row = h slice (contiguous k = 16*(lane/16) + e, same as B)
    v16bf afrag[KTILES];
    {
      const int off = (lane & 16) ? 16 : 0;
      for (int t = 0; t < KTILES; ++t) {
        union { v16bf v; __bf16 h[16]; } u;
        #pragma unroll
        for (int e = 0; e < 16; ++e) {
          int kk = t * 32 + off + e;
          u.h[e] = (__bf16)s_hpad[kk];
        }
        afrag[t] = u.v;
      }
    }
    // Each wave handles 2 N-tiles of 16 columns; B fragment = 2x 32B contiguous LDS reads.
    for (int tt = 0; tt < 2; ++tt) {
      int tile = wv + tt * 8;
      int colLocal = tile * 16 + (lane & 15);
      const char* bbase = (const char*)ldsW + (size_t)colLocal * KPAD * 2 + ((lane & 16) ? 32 : 0);
      v8f acc = {};
      #pragma unroll
      for (int t = 0; t < KTILES; ++t) {
        v16bf bfrag = *(const v16bf*)(bbase + t * 64);
        acc = __builtin_amdgcn_wmma_f32_16x16x32_bf16(
            false, afrag[t], false, bfrag, (short)0, acc, false, false);
      }
      int gcol = wg * COLS_PER_WG + colLocal;
      float lg = (gcol < VOCAB) ? (acc[0] + dense_b[gcol]) : -3.0e38f;
      if (lane < 16) s_logit[colLocal] = lg;
    }
    __syncthreads();

    // ---- local max/argmax (tie -> lowest index) + local stable expsum ----
    float v = s_logit[tid];
    int vi = wg * COLS_PER_WG + tid;
    #pragma unroll
    for (int d = 16; d > 0; d >>= 1) {
      float ov = __shfl_down(v, d);
      int   oi = __shfl_down(vi, d);
      if (ov > v || (ov == v && oi < vi)) { v = ov; vi = oi; }
    }
    if (lane == 0) { s_red[wv] = v; s_redi[wv] = vi; }
    __syncthreads();
    if (tid == 0) {
      float m = s_red[0]; int mi = s_redi[0];
      for (int w = 1; w < 8; ++w)
        if (s_red[w] > m || (s_red[w] == m && s_redi[w] < mi)) { m = s_red[w]; mi = s_redi[w]; }
      s_red[8] = m; s_redi[8] = mi;
    }
    __syncthreads();
    float lmax = s_red[8];
    float ev = expf(s_logit[tid] - lmax);
    #pragma unroll
    for (int d = 16; d > 0; d >>= 1) ev += __shfl_down(ev, d);
    __syncthreads();
    if (lane == 0) s_red[wv] = ev;
    __syncthreads();
    if (tid == 0) {
      float sm = 0.0f;
      for (int w = 0; w < 8; ++w) sm += s_red[w];
      ws_lmax[wg] = lmax; ws_lsum[wg] = sm; ws_larg[wg] = s_redi[8];
      __hip_atomic_fetch_add(&sync->arrive, 1u, __ATOMIC_RELEASE, __HIP_MEMORY_SCOPE_AGENT);
    }

    // ---- Stage 3 (leader): merge softmax stats + global argmax ----
    if (wg == 0 && tid == 0) {
      unsigned int target = (unsigned int)(k + 1) * NWG;
      while (__hip_atomic_load(&sync->arrive, __ATOMIC_ACQUIRE, __HIP_MEMORY_SCOPE_AGENT) < target)
        __builtin_amdgcn_s_sleep(1);
      __builtin_amdgcn_fence(__ATOMIC_ACQUIRE, "agent");
      float gm = -3.4e38f; int ga = 0x7fffffff;
      for (int w = 0; w < NWG; ++w) {
        float m = ws_lmax[w]; int a2 = ws_larg[w];
        if (m > gm || (m == gm && a2 < ga)) { gm = m; ga = a2; }
      }
      float gs = 0.0f;
      for (int w = 0; w < NWG; ++w) gs += ws_lsum[w] * expf(ws_lmax[w] - gm);
      sync->gmax = gm; sync->gsum = gs; sync->gaction = ga;
      __threadfence();
      __hip_atomic_store(&sync->flagA, k + 1, __ATOMIC_RELEASE, __HIP_MEMORY_SCOPE_AGENT);
    }
    if (tid == 0) {
      while (__hip_atomic_load(&sync->flagA, __ATOMIC_ACQUIRE, __HIP_MEMORY_SCOPE_AGENT) < k + 1)
        __builtin_amdgcn_s_sleep(1);
    }
    __syncthreads();
    __builtin_amdgcn_fence(__ATOMIC_ACQUIRE, "agent");

    // ---- Stage 4: write probs slice; WG0 updates action embedding + action output ----
    float gmax = sync->gmax, gsum = sync->gsum;
    int action = sync->gaction;
    {
      int gcol = wg * COLS_PER_WG + tid;
      if (gcol < VOCAB)
        out_probs[(size_t)s * VOCAB + gcol] = expf(s_logit[tid] - gmax) / gsum;
    }
    if (wg == 0) {
      if (tid == 0) out_actions[s] = (float)action;
      for (int i = tid; i < EMB; i += THREADS)
        s_a[i] = emb[(size_t)action * EMB + i];
    }
    __syncthreads();
    ++k;
  }
}

// ---------------- launcher ----------------
extern "C" void kernel_launch(void* const* d_in, const int* in_sizes, int n_in,
                              void* d_out, int out_size, void* d_ws, size_t ws_size,
                              hipStream_t stream) {
  (void)in_sizes; (void)n_in; (void)ws_size;
  const float* h_enc = (const float*)d_in[0];
  const float* emb   = (const float*)d_in[1];
  const float* Wx    = (const float*)d_in[2];
  const float* Wh    = (const float*)d_in[3];
  const float* b     = (const float*)d_in[4];
  const float* dW    = (const float*)d_in[5];
  const float* db    = (const float*)d_in[6];
  const int*   loc   = (const int*)d_in[7];
  const int*   mask  = (const int*)d_in[8];
  const int*   go    = (const int*)d_in[9];

  char* ws = (char*)d_ws;
  __bf16* wcm  = (__bf16*)(ws + OFF_WCM);
  float*  wxht = (float*)(ws + OFF_WXHT);
  float*  encf = (float*)(ws + OFF_ENC);
  float*  lmax = (float*)(ws + OFF_LMAX);
  float*  lsum = (float*)(ws + OFF_LSUM);
  int*    larg = (int*)(ws + OFF_LARG);
  float*  wsh  = (float*)(ws + OFF_H);
  SyncBlk* sync = (SyncBlk*)(ws + OFF_SYNC);

  float* outp = (float*)d_out;
  float* out_actions = outp;            // [4352]
  float* out_probs   = outp + STEPS;    // [4352][13042]

  { long long n = (long long)COLS_PAD * KPAD;
    prep_wcm_kernel<<<(int)((n + 255) / 256), 256, 0, stream>>>(dW, wcm); }
  { int n = GATES * XDIM;
    prep_wxht_kernel<<<(n + 255) / 256, 256, 0, stream>>>(Wx, Wh, wxht); }
  { int n = STEPS * HENC;
    prep_enc_kernel<<<(n + 255) / 256, 256, 0, stream>>>(h_enc, loc, encf); }
  prep_zero_kernel<<<2048, 256, 0, stream>>>(outp, (long long)out_size);
  prep_sync_kernel<<<1, 64, 0, stream>>>(sync, lmax, lsum, larg);

  hipFuncSetAttribute(reinterpret_cast<const void*>(decode_kernel),
                      hipFuncAttributeMaxDynamicSharedMemorySize, SMEM_BYTES);
  decode_kernel<<<NWG, THREADS, SMEM_BYTES, stream>>>(
      wcm, wxht, encf, emb, b, db, mask, go,
      lmax, lsum, larg, wsh, sync, out_actions, out_probs);
}
